// LFADS_Net_13391708029276
// MI455X (gfx1250) — matrix operand order repaired and checked by reading, not compiled, not run on hardware
//
#include <hip/hip_runtime.h>

// ---------------- problem constants ----------------
#define T_STEPS 200
#define BATCH   512
#define NCH     256
#define ESZ     256
#define CSZ     256
#define GLAT    128
#define USZ     16
#define CTSZ    256
#define GENSZ   512
#define FACSZ   64
#define CLIPV   5.0f
#define FBIAS   1.0f

typedef __attribute__((ext_vector_type(16))) _Float16 v16h;
typedef __attribute__((ext_vector_type(8)))  _Float16 v8h;
typedef __attribute__((ext_vector_type(8)))  float    v8f;

// A-matrix described as up to 3 consecutive K-segments (plus implicit zero pad).
struct ASeg { const _Float16* p; int len; int stride; };

__device__ __forceinline__ float sigm(float x)  { return 1.0f / (1.0f + __expf(-x)); }
__device__ __forceinline__ float clip5(float x) { return fminf(fmaxf(x, -CLIPV), CLIPV); }

// Load 8 contiguous f16 at global-k k0 for a given row; segment boundaries are
// multiples of 8, so one 16B vector load never crosses a segment.
__device__ __forceinline__ void load8seg(_Float16* dst, const ASeg* segs, int nseg,
                                         int row, int k0) {
  int base = 0;
  for (int s = 0; s < nseg; ++s) {
    int e = base + segs[s].len;
    if (k0 < e) {
      const v8h* vp = (const v8h*)(segs[s].p + (size_t)row * segs[s].stride + (k0 - base));
      v8h v = *vp;
#pragma unroll
      for (int i = 0; i < 8; ++i) dst[i] = v[i];
      return;
    }
    base = e;
  }
#pragma unroll
  for (int i = 0; i < 8; ++i) dst[i] = (_Float16)0.0f;  // zero pad region
}

// A fragment, 16x32 f16 (ISA 7.12.2): lane holds row m=lane&15;
// element j -> K = kb + (j%8) + 8*(lane>>4) + 16*(j/8)
__device__ __forceinline__ v16h loadA(const ASeg* segs, int nseg, int rowBase,
                                      int kb, int lane) {
  int m = lane & 15, hi = lane >> 4;
  _Float16 t[16];
  load8seg(t,     segs, nseg, rowBase + m, kb + 8 * hi);
  load8seg(t + 8, segs, nseg, rowBase + m, kb + 16 + 8 * hi);
  v16h a;
#pragma unroll
  for (int i = 0; i < 16; ++i) a[i] = t[i];
  return a;
}

// B fragment from weight stored [N,K] row-major (i.e. B^T): lane holds col n=lane&15;
// element j -> K = kb + (j%8) + 8*(lane>>4) + 16*(j/8): two 16B runs.
__device__ __forceinline__ v16h loadB(const _Float16* W, int K, int colBase, int kb, int lane) {
  int n = lane & 15, hi = lane >> 4;
  const v8h* vp = (const v8h*)(W + (size_t)(colBase + n) * K + kb + 8 * hi);
  v8h lo = vp[0];
  v8h hv = vp[2];
  v16h b;
#pragma unroll
  for (int i = 0; i < 8; ++i) { b[i] = lo[i]; b[i + 8] = hv[i]; }
  return b;
}

// One 16x16 output tile (used for the small tail GEMMs).
__device__ __forceinline__ v8f tile_gemm(const ASeg* segs, int nseg, int rowBase,
                                         const _Float16* W, int K, int colBase, int lane) {
  v8f acc = {0.f, 0.f, 0.f, 0.f, 0.f, 0.f, 0.f, 0.f};
  for (int kb = 0; kb < K; kb += 32) {
    v16h a = loadA(segs, nseg, rowBase, kb, lane);
    v16h b = loadB(W, K, colBase, kb, lane);
    acc = __builtin_amdgcn_wmma_f32_16x16x32_f16(false, a, false, b, (short)0, acc,
                                                 false, false);
  }
  return acc;
}

// Multi-column-tile GEMM: one wave owns NT tiles in the same 16-row band
// (colBase = (wave + 8*i)*16). A fragment loaded once per K-chunk, reused for
// all NT column tiles -> ~NT x fewer A-side vmem per WMMA.
template <int NT>
__device__ __forceinline__ void tile_gemm_multi(const ASeg* segs, int nseg, int rowBase,
                                                const _Float16* W, int K, int wave,
                                                int lane, v8f* acc) {
#pragma unroll
  for (int i = 0; i < NT; ++i) acc[i] = (v8f){0.f, 0.f, 0.f, 0.f, 0.f, 0.f, 0.f, 0.f};
  for (int kb = 0; kb < K; kb += 32) {
    v16h a = loadA(segs, nseg, rowBase, kb, lane);
#pragma unroll
    for (int i = 0; i < NT; ++i) {
      v16h b = loadB(W, K, (wave + 8 * i) * 16, kb, lane);
      acc[i] = __builtin_amdgcn_wmma_f32_16x16x32_f16(false, a, false, b, (short)0,
                                                      acc[i], false, false);
    }
  }
}

// Store D tile (+bias) into a [16][ldb] float band in LDS.
__device__ __forceinline__ void store_band(float* band, int ldb, int colBase, v8f acc,
                                           const float* bias, int lane) {
  int n = lane & 15, hi = lane >> 4;
  float bv = bias ? bias[colBase + n] : 0.0f;
#pragma unroll
  for (int r = 0; r < 8; ++r) band[(r + 8 * hi) * ldb + colBase + n] = acc[r] + bv;
}

// ---------------- utility kernels ----------------
__global__ void k_cvt(_Float16* dst, const float* src, long long n) {
  long long i = (long long)blockIdx.x * blockDim.x + threadIdx.x;
  if (i < n) dst[i] = (_Float16)src[i];
}
__global__ void k_cvt_pad(_Float16* dst, const float* src, int N, int Ks, int Kd) {
  long long i = (long long)blockIdx.x * blockDim.x + threadIdx.x;
  if (i >= (long long)N * Kd) return;
  int r = (int)(i / Kd), k = (int)(i % Kd);
  dst[i] = (k < Ks) ? (_Float16)src[(size_t)r * Ks + k] : (_Float16)0.0f;
}
__global__ void k_zero_f32(float* p, long long n) {
  long long i = (long long)blockIdx.x * blockDim.x + threadIdx.x;
  if (i < n) p[i] = 0.0f;
}
__global__ void k_zero_f16(_Float16* p, long long n) {
  long long i = (long long)blockIdx.x * blockDim.x + threadIdx.x;
  if (i < n) p[i] = (_Float16)0.0f;
}

// ---------------- phase 1: hoisted input projections ----------------
// Xp[s][t*B+b][n] = x[t,b,:] @ Wih[s]^T + bih[s].
// grid (TB/16, 1, 4); each block emits a full 16x768 band (8 waves x 6 tiles).
__global__ __launch_bounds__(256) void k_xproj(const _Float16* x16, const _Float16* Wih,
                                               const float* b0, const float* b1,
                                               const float* b2, const float* b3,
                                               float* Xp) {
  int lane = threadIdx.x & 31, wave = threadIdx.x >> 5;
  int s = blockIdx.z;
  int rowBase = blockIdx.x * 16;
  const float* bias = (s == 0) ? b0 : (s == 1) ? b1 : (s == 2) ? b2 : b3;
  ASeg seg = { x16, NCH, NCH };
  v8f acc[6];
  tile_gemm_multi<6>(&seg, 1, rowBase, Wih + (size_t)s * 768 * NCH, NCH, wave, lane, acc);
  float* out = Xp + (size_t)s * (size_t)T_STEPS * BATCH * 768;
  int n = lane & 15, hi = lane >> 4;
#pragma unroll
  for (int i = 0; i < 6; ++i) {
    int colBase = (wave + 8 * i) * 16;
    float bv = bias[colBase + n];
#pragma unroll
    for (int r = 0; r < 8; ++r)
      out[(size_t)(rowBase + r + 8 * hi) * 768 + colBase + n] = acc[i][r] + bv;
  }
}

// ---------------- phase 2: encoder recurrence (4 streams: g_f,g_b,c_f,c_b) ----------------
struct EncP {
  _Float16* h16; float* hF;
  const _Float16* Whh; const float* Xp;
  const float* bhh0; const float* bhh1; const float* bhh2; const float* bhh3;
  _Float16* outc16;
  int t;
};
__global__ __launch_bounds__(256) void k_enc_step(EncP P) {
  extern __shared__ float lds[];  // 16 x 768 floats
  int lane = threadIdx.x & 31, wave = threadIdx.x >> 5;
  int s = blockIdx.y;
  int rowBase = blockIdx.x * 16;
  _Float16* h16 = P.h16 + (size_t)s * BATCH * ESZ;
  float*    hF  = P.hF  + (size_t)s * BATCH * ESZ;
  const _Float16* Whh = P.Whh + (size_t)s * 3 * ESZ * ESZ;
  const float* bhh = (s == 0) ? P.bhh0 : (s == 1) ? P.bhh1 : (s == 2) ? P.bhh2 : P.bhh3;
  int tIdx = (s & 1) ? (T_STEPS - 1 - P.t) : P.t;   // backward streams read reversed x
  const float* Xp = P.Xp + ((size_t)s * T_STEPS * BATCH + (size_t)tIdx * BATCH) * 768;

  ASeg seg = { h16, ESZ, ESZ };
  v8f acc[6];
  tile_gemm_multi<6>(&seg, 1, rowBase, Whh, ESZ, wave, lane, acc);
#pragma unroll
  for (int i = 0; i < 6; ++i) store_band(lds, 768, (wave + 8 * i) * 16, acc[i], bhh, lane);
  __syncthreads();
  for (int idx = threadIdx.x; idx < 16 * ESZ; idx += 256) {
    int m = idx >> 8, e = idx & 255;
    int b = rowBase + m;
    const float* xp = Xp + (size_t)b * 768;
    float r  = sigm(xp[e]       + lds[m * 768 + e]);
    float z  = sigm(xp[256 + e] + lds[m * 768 + 256 + e]);
    float nn = tanhf(xp[512 + e] + r * lds[m * 768 + 512 + e]);
    float hold = hF[(size_t)b * ESZ + e];
    float hnew = (1.0f - z) * nn + z * hold;
    hF[(size_t)b * ESZ + e]  = hnew;
    h16[(size_t)b * ESZ + e] = (_Float16)hnew;
    if (s >= 2)  // 'c' encoder emits clipped outputs at every t
      P.outc16[((size_t)tIdx * BATCH + b) * 512 + ((s == 3) ? 256 : 0) + e] =
          (_Float16)clip5(hnew);
  }
}

// ---------------- phase 3: latent init ----------------
__global__ void k_build_hg(_Float16* hg16, const float* hF) {
  int i = blockIdx.x * blockDim.x + threadIdx.x;
  if (i >= BATCH * 512) return;
  int b = i >> 9, c = i & 511;
  const float* src = (c < 256) ? hF : hF + (size_t)BATCH * ESZ;
  hg16[i] = (_Float16)clip5(src[(size_t)b * ESZ + (c & 255)]);
}

__global__ __launch_bounds__(256) void k_g0(const _Float16* hg16, const _Float16* Wg0,
                                            const float* bg0, const float* eps,
                                            _Float16* g016) {
  extern __shared__ float lds[];  // 16 x 256
  int lane = threadIdx.x & 31, wave = threadIdx.x >> 5;
  int rowBase = blockIdx.x * 16;
  ASeg seg = { hg16, 512, 512 };
  v8f acc[2];
  tile_gemm_multi<2>(&seg, 1, rowBase, Wg0, 512, wave, lane, acc);
#pragma unroll
  for (int i = 0; i < 2; ++i) store_band(lds, 256, (wave + 8 * i) * 16, acc[i], bg0, lane);
  __syncthreads();
  for (int idx = threadIdx.x; idx < 16 * GLAT; idx += 256) {
    int m = idx >> 7, j = idx & 127;
    int b = rowBase + m;
    float mean = lds[m * 256 + j], logv = lds[m * 256 + 128 + j];
    g016[(size_t)b * GLAT + j] =
        (_Float16)(mean + __expf(0.5f * logv) * eps[(size_t)b * GLAT + j]);
  }
}

__global__ __launch_bounds__(256) void k_gs(const _Float16* g016, const _Float16* Wgs,
                                            const float* bgs, float* gstF, _Float16* gst16) {
  extern __shared__ float lds[];  // 16 x 512
  int lane = threadIdx.x & 31, wave = threadIdx.x >> 5;
  int rowBase = blockIdx.x * 16;
  ASeg seg = { g016, GLAT, GLAT };
  v8f acc[4];
  tile_gemm_multi<4>(&seg, 1, rowBase, Wgs, GLAT, wave, lane, acc);
#pragma unroll
  for (int i = 0; i < 4; ++i) store_band(lds, 512, (wave + 8 * i) * 16, acc[i], bgs, lane);
  __syncthreads();
  for (int idx = threadIdx.x; idx < 16 * GENSZ; idx += 256) {
    int m = idx >> 9, c = idx & 511;
    int b = rowBase + m;
    float v = lds[m * 512 + c];              // no clip on initial gen state
    gstF[(size_t)b * GENSZ + c]  = v;
    gst16[(size_t)b * GENSZ + c] = (_Float16)v;
  }
}

__global__ __launch_bounds__(256) void k_fac0(const _Float16* gst16, const _Float16* Wfac,
                                              _Float16* fst16) {
  int lane = threadIdx.x & 31, wave = threadIdx.x >> 5;
  int rowBase = blockIdx.x * 16;
  if (wave < 4) {
    ASeg seg = { gst16, GENSZ, GENSZ };
    v8f acc = tile_gemm(&seg, 1, rowBase, Wfac, GENSZ, wave * 16, lane);
    int n = lane & 15, hi = lane >> 4;
#pragma unroll
    for (int r = 0; r < 8; ++r)
      fst16[(size_t)(rowBase + r + 8 * hi) * FACSZ + wave * 16 + n] = (_Float16)acc[r];
  }
}

// ---------------- phase 4: controller + generator scan ----------------
// C1: ru = cat(enc_t, fst, cst) @ Wru^T + b ; r=sig, u=sig(+FBIAS); emit r*h (f16), u (f32)
__global__ __launch_bounds__(256) void k_ctl_ru(const _Float16* outc16, const _Float16* fst16,
                                                const _Float16* cst16, const float* cstF,
                                                const _Float16* Wru, const float* bru,
                                                _Float16* rh16, float* uF, int t) {
  extern __shared__ float lds[];  // 16 x 512
  int lane = threadIdx.x & 31, wave = threadIdx.x >> 5;
  int rowBase = blockIdx.x * 16;
  ASeg segs[3] = { { outc16 + (size_t)t * BATCH * 512, 512, 512 },
                   { fst16, FACSZ, FACSZ },
                   { cst16, CTSZ, CTSZ } };
  v8f acc[4];
  tile_gemm_multi<4>(segs, 3, rowBase, Wru, 832, wave, lane, acc);
#pragma unroll
  for (int i = 0; i < 4; ++i) store_band(lds, 512, (wave + 8 * i) * 16, acc[i], bru, lane);
  __syncthreads();
  for (int idx = threadIdx.x; idx < 16 * CTSZ; idx += 256) {
    int m = idx >> 8, e = idx & 255;
    int b = rowBase + m;
    float r = sigm(lds[m * 512 + e]);
    float u = sigm(lds[m * 512 + 256 + e] + FBIAS);
    rh16[(size_t)b * CTSZ + e] = (_Float16)(r * cstF[(size_t)b * CTSZ + e]);
    uF[(size_t)b * CTSZ + e] = u;
  }
}

// C2: c = tanh(cat(enc_t, fst, r*h) @ Wc^T + b); cst = clip(u*h + (1-u)*c);
//     then u-latent: (mean,logvar) = cst @ Wu^T + bu; sample u -> u16.
__global__ __launch_bounds__(256) void k_ctl_c(const _Float16* outc16, const _Float16* fst16,
                                               const _Float16* rh16, const _Float16* Wc,
                                               const float* bc, const float* uF,
                                               float* cstF, _Float16* cst16,
                                               const _Float16* Wu, const float* bu,
                                               const float* eps_u, _Float16* u16, int t) {
  extern __shared__ char smem[];
  float*    cband = (float*)smem;                              // 16*256 f32
  _Float16* cst_s = (_Float16*)(smem + 16 * 256 * 4);          // 16*256 f16
  float*    uband = (float*)(smem + 16 * 256 * 4 + 16 * 256 * 2); // 16*32 f32
  int lane = threadIdx.x & 31, wave = threadIdx.x >> 5;
  int rowBase = blockIdx.x * 16;
  ASeg segs[3] = { { outc16 + (size_t)t * BATCH * 512, 512, 512 },
                   { fst16, FACSZ, FACSZ },
                   { rh16, CTSZ, CTSZ } };
  v8f acc[2];
  tile_gemm_multi<2>(segs, 3, rowBase, Wc, 832, wave, lane, acc);
#pragma unroll
  for (int i = 0; i < 2; ++i) store_band(cband, 256, (wave + 8 * i) * 16, acc[i], bc, lane);
  __syncthreads();
  for (int idx = threadIdx.x; idx < 16 * CTSZ; idx += 256) {
    int m = idx >> 8, e = idx & 255;
    int b = rowBase + m;
    float c = tanhf(cband[m * 256 + e]);
    float u = uF[(size_t)b * CTSZ + e];
    float hnew = clip5(u * cstF[(size_t)b * CTSZ + e] + (1.0f - u) * c);
    cstF[(size_t)b * CTSZ + e] = hnew;
    _Float16 h = (_Float16)hnew;
    cst16[(size_t)b * CTSZ + e] = h;
    cst_s[m * 256 + e] = h;
  }
  __syncthreads();
  if (wave < 2) {                            // wave-uniform branch: EXEC stays all-ones
    ASeg seg = { cst_s, CTSZ, CTSZ };
    v8f acc1 = tile_gemm(&seg, 1, 0, Wu, CTSZ, wave * 16, lane);
    store_band(uband, 32, wave * 16, acc1, bu, lane);
  }
  __syncthreads();
  for (int idx = threadIdx.x; idx < 16 * USZ; idx += 256) {
    int m = idx >> 4, j = idx & 15;
    int b = rowBase + m;
    float mean = uband[m * 32 + j], logv = uband[m * 32 + 16 + j];
    float uv = mean + __expf(0.5f * logv) * eps_u[((size_t)t * BATCH + b) * USZ + j];
    u16[(size_t)b * USZ + j] = (_Float16)uv;
  }
}

// G1: gen GRU gates, K = 16 + 512 (+pad to 544)
__global__ __launch_bounds__(256) void k_gen_ru(const _Float16* u16, const _Float16* gst16,
                                                const float* gstF, const _Float16* Wru,
                                                const float* bru, _Float16* rh16g, float* ugF) {
  extern __shared__ float lds[];  // 16 x 1024
  int lane = threadIdx.x & 31, wave = threadIdx.x >> 5;
  int rowBase = blockIdx.x * 16;
  ASeg segs[2] = { { u16, USZ, USZ }, { gst16, GENSZ, GENSZ } };  // k>=528 -> zeros
  v8f acc[8];
  tile_gemm_multi<8>(segs, 2, rowBase, Wru, 544, wave, lane, acc);
#pragma unroll
  for (int i = 0; i < 8; ++i) store_band(lds, 1024, (wave + 8 * i) * 16, acc[i], bru, lane);
  __syncthreads();
  for (int idx = threadIdx.x; idx < 16 * GENSZ; idx += 256) {
    int m = idx >> 9, e = idx & 511;
    int b = rowBase + m;
    float r = sigm(lds[m * 1024 + e]);
    float u = sigm(lds[m * 1024 + 512 + e] + FBIAS);
    rh16g[(size_t)b * GENSZ + e] = (_Float16)(r * gstF[(size_t)b * GENSZ + e]);
    ugF[(size_t)b * GENSZ + e] = u;
  }
}

// G2: gen candidate + state update + factor projection (factors -> d_out)
__global__ __launch_bounds__(256) void k_gen_c(const _Float16* u16, const _Float16* rh16g,
                                               const _Float16* Wc, const float* bc,
                                               const float* ugF, float* gstF, _Float16* gst16,
                                               const _Float16* Wfac, float* out,
                                               _Float16* fst16, int t) {
  extern __shared__ char smem[];
  float*    cband = (float*)smem;                       // 16*512 f32
  _Float16* gst_s = (_Float16*)(smem + 16 * 512 * 4);   // 16*512 f16
  int lane = threadIdx.x & 31, wave = threadIdx.x >> 5;
  int rowBase = blockIdx.x * 16;
  ASeg segs[2] = { { u16, USZ, USZ }, { rh16g, GENSZ, GENSZ } };
  v8f acc[4];
  tile_gemm_multi<4>(segs, 2, rowBase, Wc, 544, wave, lane, acc);
#pragma unroll
  for (int i = 0; i < 4; ++i) store_band(cband, 512, (wave + 8 * i) * 16, acc[i], bc, lane);
  __syncthreads();
  for (int idx = threadIdx.x; idx < 16 * GENSZ; idx += 256) {
    int m = idx >> 9, e = idx & 511;
    int b = rowBase + m;
    float c = tanhf(cband[m * 512 + e]);
    float u = ugF[(size_t)b * GENSZ + e];
    float hnew = clip5(u * gstF[(size_t)b * GENSZ + e] + (1.0f - u) * c);
    gstF[(size_t)b * GENSZ + e] = hnew;
    _Float16 h = (_Float16)hnew;
    gst16[(size_t)b * GENSZ + e] = h;
    gst_s[m * 512 + e] = h;
  }
  __syncthreads();
  if (wave < 4) {
    ASeg seg = { gst_s, GENSZ, GENSZ };
    v8f acc1 = tile_gemm(&seg, 1, 0, Wfac, GENSZ, wave * 16, lane);
    int n = lane & 15, hi = lane >> 4;
#pragma unroll
    for (int r = 0; r < 8; ++r) {
      int b = rowBase + r + 8 * hi;
      int col = wave * 16 + n;
      out[((size_t)t * BATCH + b) * FACSZ + col] = acc1[r];
      fst16[(size_t)b * FACSZ + col] = (_Float16)acc1[r];
    }
  }
}

// ---------------- host orchestration ----------------
extern "C" void kernel_launch(void* const* d_in, const int* in_sizes, int n_in,
                              void* d_out, int out_size, void* d_ws, size_t ws_size,
                              hipStream_t stream) {
  (void)in_sizes; (void)n_in; (void)out_size; (void)ws_size;
  const float* x       = (const float*)d_in[0];
  const float* eps_g0  = (const float*)d_in[1];
  const float* eps_u   = (const float*)d_in[2];
  const float* Wg_ih_f = (const float*)d_in[3];
  const float* Wg_hh_f = (const float*)d_in[4];
  const float* bg_ih_f = (const float*)d_in[5];
  const float* bg_hh_f = (const float*)d_in[6];
  const float* Wg_ih_b = (const float*)d_in[7];
  const float* Wg_hh_b = (const float*)d_in[8];
  const float* bg_ih_b = (const float*)d_in[9];
  const float* bg_hh_b = (const float*)d_in[10];
  const float* Wc_ih_f = (const float*)d_in[11];
  const float* Wc_hh_f = (const float*)d_in[12];
  const float* bc_ih_f = (const float*)d_in[13];
  const float* bc_hh_f = (const float*)d_in[14];
  const float* Wc_ih_b = (const float*)d_in[15];
  const float* Wc_hh_b = (const float*)d_in[16];
  const float* bc_ih_b = (const float*)d_in[17];
  const float* bc_hh_b = (const float*)d_in[18];
  const float* W_g0    = (const float*)d_in[19];
  const float* b_g0    = (const float*)d_in[20];
  const float* W_gs    = (const float*)d_in[21];
  const float* b_gs    = (const float*)d_in[22];
  const float* Wctl_ru = (const float*)d_in[23];
  const float* bctl_ru = (const float*)d_in[24];
  const float* Wctl_c  = (const float*)d_in[25];
  const float* bctl_c  = (const float*)d_in[26];
  const float* W_u     = (const float*)d_in[27];
  const float* b_u     = (const float*)d_in[28];
  const float* Wgen_ru = (const float*)d_in[29];
  const float* bgen_ru = (const float*)d_in[30];
  const float* Wgen_c  = (const float*)d_in[31];
  const float* bgen_c  = (const float*)d_in[32];
  const float* W_fac   = (const float*)d_in[33];
  float* out = (float*)d_out;

  const size_t TB = (size_t)T_STEPS * BATCH;
  char* ws = (char*)d_ws;
  size_t off = 0;
  auto alloc = [&](size_t bytes) -> char* {
    char* p = ws + off;
    off += (bytes + 255) & ~(size_t)255;
    return p;
  };
  _Float16* x16    = (_Float16*)alloc(TB * NCH * 2);
  _Float16* Wih16  = (_Float16*)alloc((size_t)4 * 768 * NCH * 2);
  _Float16* Whh16  = (_Float16*)alloc((size_t)4 * 768 * NCH * 2);
  _Float16* Wg016  = (_Float16*)alloc((size_t)256 * 512 * 2);
  _Float16* Wgs16  = (_Float16*)alloc((size_t)512 * 128 * 2);
  _Float16* Wcru16 = (_Float16*)alloc((size_t)512 * 832 * 2);
  _Float16* Wcc16  = (_Float16*)alloc((size_t)256 * 832 * 2);
  _Float16* Wu16   = (_Float16*)alloc((size_t)32 * 256 * 2);
  _Float16* Wgru16 = (_Float16*)alloc((size_t)1024 * 544 * 2);
  _Float16* Wgc16  = (_Float16*)alloc((size_t)512 * 544 * 2);
  _Float16* Wfac16 = (_Float16*)alloc((size_t)64 * 512 * 2);
  float*    Xp     = (float*)alloc((size_t)4 * TB * 768 * 4);
  _Float16* outc16 = (_Float16*)alloc(TB * 512 * 2);
  float*    hF     = (float*)alloc((size_t)4 * BATCH * 256 * 4);
  _Float16* h16    = (_Float16*)alloc((size_t)4 * BATCH * 256 * 2);
  _Float16* hg16   = (_Float16*)alloc((size_t)BATCH * 512 * 2);
  _Float16* g016   = (_Float16*)alloc((size_t)BATCH * 128 * 2);
  float*    gstF   = (float*)alloc((size_t)BATCH * 512 * 4);
  _Float16* gst16  = (_Float16*)alloc((size_t)BATCH * 512 * 2);
  float*    cstF   = (float*)alloc((size_t)BATCH * 256 * 4);
  _Float16* cst16  = (_Float16*)alloc((size_t)BATCH * 256 * 2);
  _Float16* fst16  = (_Float16*)alloc((size_t)BATCH * 64 * 2);
  _Float16* rh16   = (_Float16*)alloc((size_t)BATCH * 256 * 2);
  float*    uF     = (float*)alloc((size_t)BATCH * 256 * 4);
  _Float16* rh16g  = (_Float16*)alloc((size_t)BATCH * 512 * 2);
  float*    ugF    = (float*)alloc((size_t)BATCH * 512 * 4);
  _Float16* u16    = (_Float16*)alloc((size_t)BATCH * 16 * 2);

  auto CVT = [&](_Float16* d, const float* s, long long n) {
    k_cvt<<<dim3((unsigned)((n + 255) / 256)), 256, 0, stream>>>(d, s, n);
  };
  // f32 -> f16 weight staging (B^T layout [N,K], K zero-padded where needed)
  CVT(x16, x, (long long)TB * NCH);
  CVT(Wih16 + (size_t)0 * 768 * 256, Wg_ih_f, 768 * 256);
  CVT(Wih16 + (size_t)1 * 768 * 256, Wg_ih_b, 768 * 256);
  CVT(Wih16 + (size_t)2 * 768 * 256, Wc_ih_f, 768 * 256);
  CVT(Wih16 + (size_t)3 * 768 * 256, Wc_ih_b, 768 * 256);
  CVT(Whh16 + (size_t)0 * 768 * 256, Wg_hh_f, 768 * 256);
  CVT(Whh16 + (size_t)1 * 768 * 256, Wg_hh_b, 768 * 256);
  CVT(Whh16 + (size_t)2 * 768 * 256, Wc_hh_f, 768 * 256);
  CVT(Whh16 + (size_t)3 * 768 * 256, Wc_hh_b, 768 * 256);
  CVT(Wg016, W_g0, 256 * 512);
  CVT(Wgs16, W_gs, 512 * 128);
  CVT(Wcru16, Wctl_ru, 512 * 832);
  CVT(Wcc16, Wctl_c, 256 * 832);
  CVT(Wu16, W_u, 32 * 256);
  CVT(Wfac16, W_fac, 64 * 512);
  k_cvt_pad<<<dim3((1024 * 544 + 255) / 256), 256, 0, stream>>>(Wgru16, Wgen_ru, 1024, 528, 544);
  k_cvt_pad<<<dim3((512 * 544 + 255) / 256), 256, 0, stream>>>(Wgc16, Wgen_c, 512, 528, 544);
  // zero initial states
  k_zero_f32<<<dim3((4 * BATCH * 256 + 255) / 256), 256, 0, stream>>>(hF, 4LL * BATCH * 256);
  k_zero_f16<<<dim3((4 * BATCH * 256 + 255) / 256), 256, 0, stream>>>(h16, 4LL * BATCH * 256);
  k_zero_f32<<<dim3((BATCH * 256 + 255) / 256), 256, 0, stream>>>(cstF, (long long)BATCH * 256);
  k_zero_f16<<<dim3((BATCH * 256 + 255) / 256), 256, 0, stream>>>(cst16, (long long)BATCH * 256);

  // Phase 1: time-parallel input projections (one fat WMMA GEMM, 4 streams)
  k_xproj<<<dim3((unsigned)(TB / 16), 1, 4), 256, 0, stream>>>(
      x16, Wih16, bg_ih_f, bg_ih_b, bc_ih_f, bc_ih_b, Xp);

  // Phase 2: encoder recurrences (sequential in t; 4 streams x 32 bands per launch)
  for (int t = 0; t < T_STEPS; ++t) {
    EncP P { h16, hF, Whh16, Xp, bg_hh_f, bg_hh_b, bc_hh_f, bc_hh_b, outc16, t };
    k_enc_step<<<dim3(BATCH / 16, 4), 256, 16 * 768 * 4, stream>>>(P);
  }

  // Phase 3: latent init
  k_build_hg<<<dim3((BATCH * 512 + 255) / 256), 256, 0, stream>>>(hg16, hF);
  k_g0<<<dim3(BATCH / 16), 256, 16 * 256 * 4, stream>>>(hg16, Wg016, b_g0, eps_g0, g016);
  k_gs<<<dim3(BATCH / 16), 256, 16 * 512 * 4, stream>>>(g016, Wgs16, b_gs, gstF, gst16);
  k_fac0<<<dim3(BATCH / 16), 256, 0, stream>>>(gst16, Wfac16, fst16);

  // Phase 4: controller + generator scan
  for (int t = 0; t < T_STEPS; ++t) {
    k_ctl_ru<<<dim3(BATCH / 16), 256, 16 * 512 * 4, stream>>>(
        outc16, fst16, cst16, cstF, Wcru16, bctl_ru, rh16, uF, t);
    k_ctl_c<<<dim3(BATCH / 16), 256, 16 * 256 * 4 + 16 * 256 * 2 + 16 * 32 * 4, stream>>>(
        outc16, fst16, rh16, Wcc16, bctl_c, uF, cstF, cst16, Wu16, b_u, eps_u, u16, t);
    k_gen_ru<<<dim3(BATCH / 16), 256, 16 * 1024 * 4, stream>>>(
        u16, gst16, gstF, Wgru16, bgen_ru, rh16g, ugF);
    k_gen_c<<<dim3(BATCH / 16), 256, 16 * 512 * 4 + 16 * 512 * 2, stream>>>(
        u16, rh16g, Wgc16, bgen_c, ugF, gstF, gst16, Wfac16, out, fst16, t);
  }
}
